// Attn_38809324487097
// MI455X (gfx1250) — matrix operand order repaired and checked by reading, not compiled, not run on hardware
//
#include <hip/hip_runtime.h>
#include <hip/hip_bf16.h>
#include <stdint.h>

// ---------------------------------------------------------------------------
// Problem constants (from reference): M=STATE_LEN, N=SEQ_LEN, K=HIDDEN
// ---------------------------------------------------------------------------
#define M_DIM 8192
#define N_DIM 8192
#define K_DIM 1024

// GEMM tiling: block = 256x128, 512 threads = 16 waves (4x4 wave grid),
// wave tile 64x32 (4x2 sub-tiles of 16x16), K-step 32 (bf16 WMMA depth).
// bf16x3 split precision: 3 WMMAs per sub-tile pair per K-step.
// LDS ping-pong double buffered; tiles filled by ASYNC global->LDS DMA
// (GLOBAL_LOAD_ASYNC_TO_LDS_B128, ASYNCcnt) issued at the top of each
// iteration so the fill overlaps the whole WMMA section.
#define BM 256
#define BN 128
#define BK 32
#define LDS_STRIDE 40   // 32 K elems + 8 pad (80B row stride -> conflict-free b128)

#define A_PLANE (BM * LDS_STRIDE)          // 10240 ushorts per buffer
#define B_PLANE (BN * LDS_STRIDE)          // 5120 ushorts per buffer
// [A_hi 2buf][A_lo 2buf][B_hi 2buf][B_lo 2buf]
#define GEMM_LDS_USHORTS (4 * A_PLANE + 4 * B_PLANE)   // 61440
#define GEMM_LDS_BYTES   (GEMM_LDS_USHORTS * 2)        // 122880 (< 320KB WGP LDS)

typedef __attribute__((ext_vector_type(16))) __bf16          v16bf;
typedef __attribute__((ext_vector_type(8)))  float           v8f;
typedef __attribute__((ext_vector_type(4)))  unsigned short  v4us;
typedef __attribute__((ext_vector_type(8)))  unsigned short  v8us;
typedef __attribute__((ext_vector_type(16))) unsigned short  v16us;

union BFrag { v16us u; v16bf b; };

__device__ __forceinline__ unsigned short f32_to_bf16_rne(float x) {
    uint32_t u = __float_as_uint(x);
    uint32_t r = u + 0x7FFFu + ((u >> 16) & 1u);   // round-to-nearest-even
    return (unsigned short)(r >> 16);
}
__device__ __forceinline__ float bf16_bits_to_f32(unsigned short h) {
    return __uint_as_float(((uint32_t)h) << 16);
}

// ---------------------------------------------------------------------------
// CDNA5 async global->LDS copy (16B per lane), tracked by ASYNCcnt.
// saddr form: mem_addr = SGPR64 + VGPR_I32 offset. LDS dest address is the
// low 32 bits of the generic pointer (ISA 10.2: LDS aperture uses addr[31:0]).
// ---------------------------------------------------------------------------
__device__ __forceinline__ void async_b128(unsigned short* lds_dst,
                                           const unsigned short* gbase,
                                           unsigned voff_bytes) {
    unsigned lds = (unsigned)(uintptr_t)lds_dst;
    asm volatile("global_load_async_to_lds_b128 %0, %1, %2"
                 :: "v"(lds), "v"(voff_bytes),
                    "s"((uint64_t)(uintptr_t)gbase)
                 : "memory");
}
__device__ __forceinline__ void wait_async0() {
    asm volatile("s_wait_asynccnt 0x0" ::: "memory");
}

// ---------------------------------------------------------------------------
// Kernel 1: split f32 -> (hi, lo) bf16 planes. One-time cost, keeps the GEMM
// hot loop free of conversion VALU work.
// ---------------------------------------------------------------------------
__global__ __launch_bounds__(256)
void split_bf16(const float* __restrict__ src,
                unsigned short* __restrict__ hi,
                unsigned short* __restrict__ lo, int n4) {
    int i = blockIdx.x * 256 + threadIdx.x;
    if (i >= n4) return;
    float4 v = ((const float4*)src)[i];
    float f[4] = {v.x, v.y, v.z, v.w};
    v4us h, l;
#pragma unroll
    for (int j = 0; j < 4; ++j) {
        unsigned short hb = f32_to_bf16_rne(f[j]);
        h[j] = hb;
        l[j] = f32_to_bf16_rne(f[j] - bf16_bits_to_f32(hb));
    }
    *(v4us*)(hi + (size_t)i * 4) = h;   // 8-byte stores
    *(v4us*)(lo + (size_t)i * 4) = l;
}

// ---------------------------------------------------------------------------
// Kernel 2: C[i,j] = sum_k A[i,k] * B[j,k] with bf16x3 split precision.
// ---------------------------------------------------------------------------
__global__ __launch_bounds__(512)
void gemm_bf16x3(const unsigned short* __restrict__ Ahi,
                 const unsigned short* __restrict__ Alo,
                 const unsigned short* __restrict__ Bhi,
                 const unsigned short* __restrict__ Blo,
                 float* __restrict__ C) {
    extern __shared__ unsigned short smem[];
    unsigned short* const sAhi = smem;                       // 2 x A_PLANE
    unsigned short* const sAlo = smem + 2 * A_PLANE;         // 2 x A_PLANE
    unsigned short* const sBhi = smem + 4 * A_PLANE;         // 2 x B_PLANE
    unsigned short* const sBlo = smem + 4 * A_PLANE + 2 * B_PLANE;

    const int tid     = threadIdx.x;
    const int lane    = tid & 31;
    const int wave    = tid >> 5;          // 0..15
    const int waveM   = wave & 3;          // 4 waves tile M (64 rows each)
    const int waveN   = wave >> 2;         // 4 waves tile N (32 cols each)
    const int blockM  = blockIdx.y * BM;
    const int blockN  = blockIdx.x * BN;
    const int halfSel = lane >> 4;         // 0: lanes 0-15, 1: lanes 16-31
    const int l15     = lane & 15;

    // Cooperative fill mapping (8-elem = 16B chunks):
    // A plane: 256 rows x 4 chunks = 1024 chunks; each thread does tid, tid+512.
    // B plane: 128 rows x 4 chunks =  512 chunks; each thread does tid.
    const int arow = tid >> 2;             // 0..127 (second chunk is arow+128)
    const int akc  = (tid & 3) * 8;
    const int brow = tid >> 2;
    const int bkc  = (tid & 3) * 8;

    // per-lane LDS chunk slots (buffer 0; buffer 1 adds plane offset)
    unsigned short* const ldsA0 = sAhi + arow * LDS_STRIDE + akc;
    unsigned short* const ldsA1 = sAhi + (arow + 128) * LDS_STRIDE + akc;
    unsigned short* const ldsAL0 = sAlo + arow * LDS_STRIDE + akc;
    unsigned short* const ldsAL1 = sAlo + (arow + 128) * LDS_STRIDE + akc;
    unsigned short* const ldsB0 = sBhi + brow * LDS_STRIDE + bkc;
    unsigned short* const ldsBL0 = sBlo + brow * LDS_STRIDE + bkc;

    // Issue the 6 async DMA fills for K-step kb into buffer `buf`.
    auto asyncLoadTile = [&](int kb, int buf) {
        unsigned offA0 = (unsigned)(((blockM + arow) * K_DIM + kb + akc) * 2);
        unsigned offA1 = offA0 + (unsigned)(128 * K_DIM * 2);
        unsigned offB  = (unsigned)(((blockN + brow) * K_DIM + kb + bkc) * 2);
        const int ab = buf * A_PLANE, bb = buf * B_PLANE;
        async_b128(ldsA0  + ab, Ahi, offA0);
        async_b128(ldsA1  + ab, Ahi, offA1);
        async_b128(ldsAL0 + ab, Alo, offA0);
        async_b128(ldsAL1 + ab, Alo, offA1);
        async_b128(ldsB0  + bb, Bhi, offB);
        async_b128(ldsBL0 + bb, Blo, offB);
    };

    v8f acc[4][2] = {};                    // 4x2 tiles of 16x16 f32 per wave

    // A-fragment loader: ISA layout for 16-bit A 16x32:
    // lane<16 holds K 0..7 then 16..23 of row M=lane; lane>=16 K 8..15, 24..31.
    auto loadAFrag = [&](const unsigned short* a_hi, const unsigned short* a_lo,
                         int mt, BFrag& aH, BFrag& aL) {
        int row = waveM * 64 + mt * 16 + l15;
        int k0  = halfSel * 8;
        const unsigned short* ph = a_hi + row * LDS_STRIDE + k0;
        const unsigned short* pl = a_lo + row * LDS_STRIDE + k0;
        v8us h0 = *(const v8us*)(ph);
        v8us h1 = *(const v8us*)(ph + 16);
        v8us l0 = *(const v8us*)(pl);
        v8us l1 = *(const v8us*)(pl + 16);
        aH.u = __builtin_shufflevector(h0, h1, 0,1,2,3,4,5,6,7,8,9,10,11,12,13,14,15);
        aL.u = __builtin_shufflevector(l0, l1, 0,1,2,3,4,5,6,7,8,9,10,11,12,13,14,15);
    };

    auto computeTile = [&](int buf) {
        const unsigned short* a_hi = sAhi + buf * A_PLANE;
        const unsigned short* a_lo = sAlo + buf * A_PLANE;
        const unsigned short* b_hi = sBhi + buf * B_PLANE;
        const unsigned short* b_lo = sBlo + buf * B_PLANE;

        // B fragments, kept live across all mt. ISA layout for 16-bit B 32x16:
        // lane<16 holds K 0..7 then 16..23 of col N=lane; lane>=16 K 8..15, 24..31.
        BFrag bH[2], bL[2];
#pragma unroll
        for (int nt = 0; nt < 2; ++nt) {
            int col = waveN * 32 + nt * 16 + l15;
            int k0  = halfSel * 8;
            const unsigned short* ph = b_hi + col * LDS_STRIDE + k0;
            const unsigned short* pl = b_lo + col * LDS_STRIDE + k0;
            v8us h0 = *(const v8us*)(ph);
            v8us h1 = *(const v8us*)(ph + 16);
            v8us l0 = *(const v8us*)(pl);
            v8us l1 = *(const v8us*)(pl + 16);
            bH[nt].u = __builtin_shufflevector(h0, h1, 0,1,2,3,4,5,6,7,8,9,10,11,12,13,14,15);
            bL[nt].u = __builtin_shufflevector(l0, l1, 0,1,2,3,4,5,6,7,8,9,10,11,12,13,14,15);
        }

        // A fragments: double-buffered stream over mt. The 6 WMMAs of
        // fragment mt overlap the ds_loads of fragment mt+1.
        BFrag aH[2], aL[2];
        loadAFrag(a_hi, a_lo, 0, aH[0], aL[0]);
#pragma unroll
        for (int mt = 0; mt < 4; ++mt) {
            const int cb = mt & 1;
            if (mt < 3) loadAFrag(a_hi, a_lo, mt + 1, aH[cb ^ 1], aL[cb ^ 1]);
            // alternate accumulator destinations between consecutive WMMAs
            acc[mt][0] = __builtin_amdgcn_wmma_f32_16x16x32_bf16(
                false, aH[cb].b, false, bH[0].b, (short)0, acc[mt][0], false, false);
            acc[mt][1] = __builtin_amdgcn_wmma_f32_16x16x32_bf16(
                false, aH[cb].b, false, bH[1].b, (short)0, acc[mt][1], false, false);
            acc[mt][0] = __builtin_amdgcn_wmma_f32_16x16x32_bf16(
                false, aH[cb].b, false, bL[0].b, (short)0, acc[mt][0], false, false);
            acc[mt][1] = __builtin_amdgcn_wmma_f32_16x16x32_bf16(
                false, aH[cb].b, false, bL[1].b, (short)0, acc[mt][1], false, false);
            acc[mt][0] = __builtin_amdgcn_wmma_f32_16x16x32_bf16(
                false, aL[cb].b, false, bH[0].b, (short)0, acc[mt][0], false, false);
            acc[mt][1] = __builtin_amdgcn_wmma_f32_16x16x32_bf16(
                false, aL[cb].b, false, bH[1].b, (short)0, acc[mt][1], false, false);
        }
    };

    // ---- prologue: DMA tile 0, wait, publish ----
    asyncLoadTile(0, 0);
    wait_async0();
    __syncthreads();

    // ---- steady state: branch-free body, last iteration peeled ----
    int cur = 0;
    for (int kb = 0; kb < K_DIM - BK; kb += BK) {
        asyncLoadTile(kb + BK, cur ^ 1);   // DMA next tile; lands under WMMAs
        computeTile(cur);
        wait_async0();                     // own async fills to nxt complete
        __syncthreads();                   // single barrier per K-step
        cur ^= 1;
    }
    computeTile(cur);                      // epilogue: no prefetch

    // ---- writeout: C/D layout: VGPR r -> M = r + 8*halfSel, N = lane&15 ----
#pragma unroll
    for (int mt = 0; mt < 4; ++mt) {
#pragma unroll
        for (int nt = 0; nt < 2; ++nt) {
            int m0 = blockM + waveM * 64 + mt * 16 + halfSel * 8;
            int n  = blockN + waveN * 32 + nt * 16 + l15;
#pragma unroll
            for (int r = 0; r < 8; ++r) {
                C[(size_t)(m0 + r) * N_DIM + n] = acc[mt][nt][r];
            }
        }
    }
}

// ---------------------------------------------------------------------------
// Kernel 3: in-place row softmax over 8192 f32 per row; row cached in LDS.
// ---------------------------------------------------------------------------
__global__ __launch_bounds__(256)
void softmax_rows(float* __restrict__ out) {
    __shared__ float row[N_DIM];           // 32 KB
    __shared__ float red[8];

    const int tid = threadIdx.x;
    float* p = out + (size_t)blockIdx.x * N_DIM;

    // load + running max
    float m = -3.402823466e38f;
    for (int i = tid * 4; i < N_DIM; i += 256 * 4) {
        float4 v = *(const float4*)(p + i);
        *(float4*)(&row[i]) = v;
        m = fmaxf(m, fmaxf(fmaxf(v.x, v.y), fmaxf(v.z, v.w)));
    }
#pragma unroll
    for (int off = 16; off > 0; off >>= 1) m = fmaxf(m, __shfl_xor(m, off));
    if ((tid & 31) == 0) red[tid >> 5] = m;
    __syncthreads();
    m = red[0];
#pragma unroll
    for (int w = 1; w < 8; ++w) m = fmaxf(m, red[w]);
    __syncthreads();                        // red will be reused

    // exp + running sum
    float s = 0.0f;
    for (int i = tid * 4; i < N_DIM; i += 256 * 4) {
        float4 v = *(float4*)(&row[i]);
        v.x = __expf(v.x - m);
        v.y = __expf(v.y - m);
        v.z = __expf(v.z - m);
        v.w = __expf(v.w - m);
        *(float4*)(&row[i]) = v;
        s += (v.x + v.y) + (v.z + v.w);
    }
#pragma unroll
    for (int off = 16; off > 0; off >>= 1) s += __shfl_xor(s, off);
    if ((tid & 31) == 0) red[tid >> 5] = s;
    __syncthreads();
    s = 0.0f;
#pragma unroll
    for (int w = 0; w < 8; ++w) s += red[w];
    const float inv = 1.0f / s;

    // normalize + store
    for (int i = tid * 4; i < N_DIM; i += 256 * 4) {
        float4 v = *(float4*)(&row[i]);
        v.x *= inv; v.y *= inv; v.z *= inv; v.w *= inv;
        *(float4*)(p + i) = v;
    }
}

// ---------------------------------------------------------------------------
// Launch: split planes into d_ws (64 MB), GEMM into d_out, softmax in place.
// ---------------------------------------------------------------------------
extern "C" void kernel_launch(void* const* d_in, const int* in_sizes, int n_in,
                              void* d_out, int out_size, void* d_ws, size_t ws_size,
                              hipStream_t stream) {
    const float* A = (const float*)d_in[0];   // out_state [8192,1024] f32
    const float* H = (const float*)d_in[1];   // history   [8192,1024] f32
    float* out = (float*)d_out;               // [8192,8192] f32

    unsigned short* ws = (unsigned short*)d_ws;
    const size_t plane = (size_t)M_DIM * K_DIM;   // 8M elements per plane
    unsigned short* Ahi = ws;
    unsigned short* Alo = ws + plane;
    unsigned short* Bhi = ws + 2 * plane;
    unsigned short* Blo = ws + 3 * plane;         // total 64 MB of d_ws

    const int n4 = (int)(plane / 4);              // float4 chunks per matrix
    split_bf16<<<(n4 + 255) / 256, 256, 0, stream>>>(A, Ahi, Alo, n4);
    split_bf16<<<(n4 + 255) / 256, 256, 0, stream>>>(H, Bhi, Blo, n4);

    dim3 grid(N_DIM / BN, M_DIM / BM);            // (64, 32)
    gemm_bf16x3<<<grid, 512, GEMM_LDS_BYTES, stream>>>(Ahi, Alo, Bhi, Blo, out);

    softmax_rows<<<M_DIM, 256, 0, stream>>>(out);
}
